// Mol2NumNet_regressor_77283641524382
// MI455X (gfx1250) — matrix operand order repaired and checked by reading, not compiled, not run on hardware
//
#include <hip/hip_runtime.h>
#include <math.h>

// ---------------------------------------------------------------------------
// Problem constants (shapes fixed by the reference)
// ---------------------------------------------------------------------------
constexpr int NN  = 40000;    // nodes
constexpr int NE  = 100000;   // edges
constexpr int D   = 64;       // hidden dim
constexpr int BAT = 256;      // batch (graphs)

// ---------------------------------------------------------------------------
// CDNA5 WMMA types / helpers (wave32, 16x16x32 bf16 -> f32)
// ---------------------------------------------------------------------------
typedef __attribute__((ext_vector_type(16))) __bf16 v16bf;
typedef __attribute__((ext_vector_type(8)))  __bf16 v8bf;
typedef __attribute__((ext_vector_type(8)))  float  v8f;
typedef __attribute__((ext_vector_type(4)))  unsigned u32x4;
typedef __attribute__((ext_vector_type(4)))  int      i32x4;
typedef __attribute__((ext_vector_type(8)))  int      i32x8;

// Tensor Data Mover path available? (this toolchain: 6-arg builtin)
#if __has_builtin(__builtin_amdgcn_tensor_load_to_lds) && __has_builtin(__builtin_amdgcn_s_wait_tensorcnt)
#define USE_TDM 1
#else
#define USE_TDM 0
#endif

__device__ inline __bf16 f2bf(float f) {
  unsigned u = __builtin_bit_cast(unsigned, f);
  unsigned r = (u + 0x7FFFu + ((u >> 16) & 1u)) >> 16;   // RNE
  return __builtin_bit_cast(__bf16, (unsigned short)r);
}

__device__ inline v8f wmma_bf16(v16bf a, v16bf b, v8f c) {
  return __builtin_amdgcn_wmma_f32_16x16x32_bf16(
      /*neg_a=*/false, a, /*neg_b=*/false, b,
      /*c_mod=*/(short)0, c, /*reuse_a=*/false, /*reuse_b=*/false);
}

// A fragment: 16x32 bf16, source row-major (row = lane&15).
// lane half (lane>>4) selects K groups {h*8..h*8+7} and {16+h*8..}.
__device__ inline v16bf load_a16(const __bf16* p, int half) {
  v8bf lo = *(const v8bf*)(p + half * 8);
  v8bf hi = *(const v8bf*)(p + 16 + half * 8);
  v16bf r;
#pragma unroll
  for (int i = 0; i < 8; ++i) { r[i] = lo[i]; r[i + 8] = hi[i]; }
  return r;
}

// B fragment: 32x16 bf16 from B^T stored N-major (row = col n = lane&15),
// lane half selects K groups {h*8..h*8+7} and {16+h*8..}. p points at &Bt[n][k_chunk_base].
__device__ inline v16bf load_b16(const __bf16* p, int half) {
  v8bf lo = *(const v8bf*)(p + half * 8);
  v8bf hi = *(const v8bf*)(p + 16 + half * 8);
  v16bf r;
#pragma unroll
  for (int i = 0; i < 8; ++i) { r[i] = lo[i]; r[i + 8] = hi[i]; }
  return r;
}

// Build a bf16 A fragment from an fp32 row (length 64), chunk kk in {0,1}
__device__ inline v16bf a_frag_f32(const float* __restrict__ row, int half, int kk) {
  v16bf r;
  const int k0 = kk * 32 + half * 8;
#pragma unroll
  for (int i = 0; i < 8; ++i) {
    r[i]     = f2bf(row[k0 + i]);
    r[i + 8] = f2bf(row[k0 + 16 + i]);
  }
  return r;
}

__device__ inline float sigm(float x) { return 1.0f / (1.0f + expf(-x)); }

// ordered encode/decode for float atomic-max on unsigned
__device__ inline unsigned fenc(float f) {
  unsigned u = __builtin_bit_cast(unsigned, f);
  return (u & 0x80000000u) ? ~u : (u | 0x80000000u);
}
__device__ inline float fdec(unsigned u) {
  unsigned v = (u & 0x80000000u) ? (u & 0x7FFFFFFFu) : ~u;
  return __builtin_bit_cast(float, v);
}

#if USE_TDM
// Issue one TDM descriptor: load a 64x64 bf16 tile (8KB, row-major, stride 64)
// from global into LDS at lds_dst. D# layout per CDNA5 ISA 8.3/8.4:
//   group0: [1:0]count=1 | lds_addr | global_addr[56:0] | type=2
//   group1: data_size=2B, tensor_dim0/1=64, tile_dim0/1=64, dim0_stride=64
// LDS byte offset = low 32 bits of the generic pointer (flat-aperture rule).
__device__ inline void tdm_load_tile64(const __bf16* gsrc, __bf16* lds_dst) {
  unsigned long long ga = (unsigned long long)(uintptr_t)gsrc;
  unsigned lds_off = (unsigned)(uintptr_t)lds_dst;
  u32x4 g0;
  g0[0] = 1u;                                              // count=1, user D#
  g0[1] = lds_off;                                         // lds_addr
  g0[2] = (unsigned)(ga & 0xffffffffull);                  // global_addr[31:0]
  g0[3] = (unsigned)((ga >> 32) & 0x01ffffffull) | (2u << 30); // addr[56:32]|type=2
  i32x8 g1;
  g1[0] = (int)(1u << 16);        // workgroup_mask=0, data_size=1 (2 bytes)
  g1[1] = (int)(64u << 16);       // atomic_barrier_addr=0, tensor_dim0[15:0]=64
  g1[2] = (int)(64u << 16);       // tensor_dim0[31:16]=0, tensor_dim1[15:0]=64
  g1[3] = (int)(64u << 16);       // tensor_dim1[31:16]=0, tile_dim0=64
  g1[4] = 64;                     // tile_dim1=64, tile_dim2=0
  g1[5] = 64;                     // tensor_dim0_stride[31:0]=64
  g1[6] = 0;                      // stride0[47:32]=0, stride1[15:0]=0
  g1[7] = 0;
  i32x4 z4 = {0, 0, 0, 0};                 // groups 2/3 unused (2-D tensor)
  i32x8 z8 = {0, 0, 0, 0, 0, 0, 0, 0};     // extra group slot (6-arg form)
  __builtin_amdgcn_tensor_load_to_lds(g0, g1, z4, z4, z8, 0);
}
#endif

// ---------------------------------------------------------------------------
// Small utility kernels
// ---------------------------------------------------------------------------
__global__ void k_f32_to_bf16(const float* __restrict__ in, __bf16* __restrict__ out, int n) {
  int i = blockIdx.x * blockDim.x + threadIdx.x;
  if (i < n) out[i] = f2bf(in[i]);
}

__global__ void k_zero_f32(float* __restrict__ p, int n) {
  int i = blockIdx.x * blockDim.x + threadIdx.x;
  if (i < n) p[i] = 0.0f;
}

// out = n_feat @ lin0_w^T + b  (K=23, VALU; weights in LDS)
__global__ void k_lin0(const float* __restrict__ nf, const float* __restrict__ w,
                       const float* __restrict__ b, float* __restrict__ out_f,
                       __bf16* __restrict__ out_b) {
  __shared__ float sw[64 * 23];
  __shared__ float sb[64];
  for (int i = threadIdx.x; i < 64 * 23; i += blockDim.x) sw[i] = w[i];
  for (int i = threadIdx.x; i < 64; i += blockDim.x) sb[i] = b[i];
  __syncthreads();
  int gid = blockIdx.x * blockDim.x + threadIdx.x;       // N*64 threads
  int n = gid >> 6, f = gid & 63;
  const float* row = nf + (size_t)n * 23;
  float acc = sb[f];
#pragma unroll
  for (int k = 0; k < 23; ++k) acc += row[k] * sw[f * 23 + k];
  out_f[gid] = acc;
  out_b[gid] = f2bf(acc);
}

// h_edge = relu(e_feat @ en_w1^T + b1)  (K=19, VALU), store bf16
__global__ void k_edge_l1(const float* __restrict__ ef, const float* __restrict__ w,
                          const float* __restrict__ b, __bf16* __restrict__ he) {
  __shared__ float sw[128 * 19];
  __shared__ float sb[128];
  for (int i = threadIdx.x; i < 128 * 19; i += blockDim.x) sw[i] = w[i];
  for (int i = threadIdx.x; i < 128; i += blockDim.x) sb[i] = b[i];
  __syncthreads();
  int gid = blockIdx.x * blockDim.x + threadIdx.x;       // E*128 threads
  int e = gid >> 7, c = gid & 127;
  const float* row = ef + (size_t)e * 19;
  float acc = sb[c];
#pragma unroll
  for (int k = 0; k < 19; ++k) acc += row[k] * sw[c * 19 + k];
  he[gid] = f2bf(fmaxf(acc, 0.0f));
}

// ---------------------------------------------------------------------------
// Big WMMA GEMM: ew[E,4096] = h_edge[E,128] @ en_w2[4096,128]^T + b2,
// stored TRANSPOSED per edge:  ewt[e][f][d]  (so message B-frags are contiguous)
// block = 128 (4 waves); wave -> one 16x16 output tile; K=128 in 4 WMMA steps
// ---------------------------------------------------------------------------
__global__ void k_edge_gemm(const __bf16* __restrict__ he,   // E x 128
                            const __bf16* __restrict__ w2,   // 4096 x 128 (N x K)
                            const float* __restrict__ b2,    // 4096
                            __bf16* __restrict__ ewt) {      // E x 64 x 64 [e][f][d]
  const int lane = threadIdx.x & 31;
  const int wv   = threadIdx.x >> 5;
  const int half = lane >> 4;
  const int col  = lane & 15;
  const int m0 = blockIdx.x * 16;                 // edge tile
  const int n0 = (blockIdx.y * 4 + wv) * 16;      // j tile (j = d*64+f)

  const __bf16* arow = he + (size_t)(m0 + col) * 128;
  const __bf16* brow = w2 + (size_t)(n0 + col) * 128;

  v8f acc = {};
#pragma unroll
  for (int kk = 0; kk < 4; ++kk) {
    v16bf a = load_a16(arow + kk * 32, half);
    v16bf b = load_b16(brow + kk * 32, half);
    acc = wmma_bf16(a, b, acc);
  }

  const int n = n0 + col;            // global j
  const int d = n >> 6, f = n & 63;  // j = d*64 + f
  const float bias = b2[n];
#pragma unroll
  for (int v = 0; v < 8; ++v) {
    const int m = m0 + v + 8 * half;
    ewt[(size_t)m * 4096 + (size_t)f * 64 + d] = f2bf(acc[v] + bias);
  }
}

// m[n][f] = conv_b[f]  (init for segment-sum)
__global__ void k_msg_init(const float* __restrict__ cb, float* __restrict__ m) {
  int gid = blockIdx.x * blockDim.x + threadIdx.x;
  m[gid] = cb[gid & 63];
}

// ---------------------------------------------------------------------------
// Message step: per edge  msg = x[src] (1x64) @ W_e (64x64), atomic-add to m[dst].
// One wave per edge iteration; x replicated into all 16 A rows; 8 WMMAs/edge.
// Memory-bound on streaming W_e (bf16, 8KB/edge). With TDM: double-buffered
// tensor_load_to_lds per wave (2 waves/block, 2 bufs * 8KB each = 32KB LDS),
// WMMA B-fragments served by ds_load; s_wait_tensorcnt paces the pipeline.
// ---------------------------------------------------------------------------
__global__ void __launch_bounds__(64)
k_message(const __bf16* __restrict__ xb,   // N x 64 bf16 (out/h)
          const __bf16* __restrict__ ewt,  // E x 64 x 64 [e][f][d]
          const int* __restrict__ src, const int* __restrict__ dst,
          float* __restrict__ m) {
  const int lane = threadIdx.x & 31;
  const int wv   = threadIdx.x >> 5;           // 0..1
  const int half = lane >> 4;
  const int col  = lane & 15;
  const size_t stride = (size_t)gridDim.x * 2;
  const size_t e0 = (size_t)blockIdx.x * 2 + wv;

#if USE_TDM
  __shared__ __bf16 sW[2][2][64 * 64];         // [buf][wave][tile]  (32 KB)
  __bf16* buf0 = &sW[0][wv][0];
  __bf16* buf1 = &sW[1][wv][0];
  if (e0 < (size_t)NE) tdm_load_tile64(ewt + e0 * 4096, buf0);
  int cur = 0;
  for (size_t e = e0; e < (size_t)NE; e += stride) {
    const size_t en = e + stride;
    if (en < (size_t)NE) {                     // start next tile DMA
      tdm_load_tile64(ewt + en * 4096, cur ? buf0 : buf1);
      __builtin_prefetch(xb + (size_t)src[en] * 64, 0, 1);
      __builtin_amdgcn_s_wait_tensorcnt(1);    // current tile landed, next in flight
    } else {
      __builtin_amdgcn_s_wait_tensorcnt(0);
    }
    const __bf16* wbase = cur ? buf1 : buf0;

    const int s  = src[e];
    const int dn = dst[e];
    const __bf16* xrow = xb + (size_t)s * 64;  // same for all lanes (replicated A)
    v16bf a0 = load_a16(xrow, half);           // K 0..31
    v16bf a1 = load_a16(xrow + 32, half);      // K 32..63

    v8f acc[4];
#pragma unroll
    for (int nt = 0; nt < 4; ++nt) {
      const __bf16* brow = wbase + (size_t)(nt * 16 + col) * 64;
      v8f c = {};
      c = wmma_bf16(a0, load_b16(brow, half), c);
      c = wmma_bf16(a1, load_b16(brow + 32, half), c);
      acc[nt] = c;
    }

    if (lane < 16) {                           // row 0 of replicated tile
      float* mrow = m + (size_t)dn * 64;
#pragma unroll
      for (int nt = 0; nt < 4; ++nt)
        atomicAdd(mrow + nt * 16 + lane, acc[nt][0]);
    }
    cur ^= 1;
  }
#else
  for (size_t e = e0; e < (size_t)NE; e += stride) {
    const int s  = src[e];
    const int dn = dst[e];
    const __bf16* xrow = xb + (size_t)s * 64;
    v16bf a0 = load_a16(xrow, half);
    v16bf a1 = load_a16(xrow + 32, half);
    const __bf16* wbase = ewt + (size_t)e * 4096;

    v8f acc[4];
#pragma unroll
    for (int nt = 0; nt < 4; ++nt) {
      const __bf16* brow = wbase + (size_t)(nt * 16 + col) * 64;
      v8f c = {};
      c = wmma_bf16(a0, load_b16(brow, half), c);
      c = wmma_bf16(a1, load_b16(brow + 32, half), c);
      acc[nt] = c;
    }

    const size_t en = e + stride;
    if (en < (size_t)NE) {
      __builtin_prefetch(ewt + en * 4096, 0, 1);
      __builtin_prefetch(xb + (size_t)src[en] * 64, 0, 1);
    }

    if (lane < 16) {
      float* mrow = m + (size_t)dn * 64;
#pragma unroll
      for (int nt = 0; nt < 4; ++nt)
        atomicAdd(mrow + nt * 16 + lane, acc[nt][0]);
    }
  }
#endif
}

// ---------------------------------------------------------------------------
// Fused GRU step. One wave per 16-node tile. 48 WMMAs/wave.
// gi = m@Wih^T+bih ; gh = h@Whh^T+bhh ; gates in VALU; writes h (f32 + bf16).
// ---------------------------------------------------------------------------
__device__ inline v8f tile_pair(v16bf a0, v16bf a1, const __bf16* wrow, int half) {
  v8f acc = {};
  acc = wmma_bf16(a0, load_b16(wrow, half), acc);
  acc = wmma_bf16(a1, load_b16(wrow + 32, half), acc);
  return acc;
}

__global__ void k_gru(const float* __restrict__ mbuf,          // N x 64
                      float* __restrict__ h,                   // N x 64 (in/out)
                      __bf16* __restrict__ hb,                 // N x 64 bf16 (out)
                      const __bf16* __restrict__ wih,          // 192 x 64 bf16
                      const __bf16* __restrict__ whh,          // 192 x 64 bf16
                      const float* __restrict__ bih, const float* __restrict__ bhh,
                      int ntiles) {
  const int lane = threadIdx.x & 31;
  const int wv   = threadIdx.x >> 5;
  const int half = lane >> 4;
  const int col  = lane & 15;
  const int tile = blockIdx.x * 8 + wv;
  if (tile >= ntiles) return;
  const int m0 = tile * 16;

  const float* mrow = mbuf + (size_t)(m0 + col) * 64;
  const float* hrow = h    + (size_t)(m0 + col) * 64;
  v16bf am0 = a_frag_f32(mrow, half, 0), am1 = a_frag_f32(mrow, half, 1);
  v16bf ah0 = a_frag_f32(hrow, half, 0), ah1 = a_frag_f32(hrow, half, 1);

  v8f rr[4], zz[4], nn[4];
#pragma unroll
  for (int ft = 0; ft < 4; ++ft) {
    {   // r gate rows [0,64)
      const int nb = 0 * 64 + ft * 16;
      v8f gi = tile_pair(am0, am1, wih + (size_t)(nb + col) * 64, half);
      v8f gh = tile_pair(ah0, ah1, whh + (size_t)(nb + col) * 64, half);
      const float bb = bih[nb + col] + bhh[nb + col];
#pragma unroll
      for (int v = 0; v < 8; ++v) rr[ft][v] = sigm(gi[v] + gh[v] + bb);
    }
    {   // z gate rows [64,128)
      const int nb = 1 * 64 + ft * 16;
      v8f gi = tile_pair(am0, am1, wih + (size_t)(nb + col) * 64, half);
      v8f gh = tile_pair(ah0, ah1, whh + (size_t)(nb + col) * 64, half);
      const float bb = bih[nb + col] + bhh[nb + col];
#pragma unroll
      for (int v = 0; v < 8; ++v) zz[ft][v] = sigm(gi[v] + gh[v] + bb);
    }
  }
#pragma unroll
  for (int ft = 0; ft < 4; ++ft) {   // n gate rows [128,192)
    const int nb = 2 * 64 + ft * 16;
    v8f gi = tile_pair(am0, am1, wih + (size_t)(nb + col) * 64, half);
    v8f gh = tile_pair(ah0, ah1, whh + (size_t)(nb + col) * 64, half);
    const float bi = bih[nb + col], bh = bhh[nb + col];
#pragma unroll
    for (int v = 0; v < 8; ++v)
      nn[ft][v] = tanhf(gi[v] + bi + rr[ft][v] * (gh[v] + bh));
  }
  // h_new = (1-z)*n + z*h
#pragma unroll
  for (int ft = 0; ft < 4; ++ft) {
#pragma unroll
    for (int v = 0; v < 8; ++v) {
      const int row = m0 + v + 8 * half;
      const int cj  = ft * 16 + col;
      const size_t idx = (size_t)row * 64 + cj;
      const float hold = h[idx];
      const float hn = (1.0f - zz[ft][v]) * nn[ft][v] + zz[ft][v] * hold;
      h[idx] = hn;
      hb[idx] = f2bf(hn);
    }
  }
}

// ---------------------------------------------------------------------------
// Set2Set kernels (B=256, D=64 — tiny, VALU)
// ---------------------------------------------------------------------------
// g[b][j] = bih[j]+bhh[j] + x@wih_row + h@whh_row.  mode: 2=x==0, 1=x=[xa|xb] (128), 0=x=xa (64)
__global__ void k_lstm_gemm(const float* __restrict__ xa, const float* __restrict__ xb,
                            const float* __restrict__ h,
                            const float* __restrict__ wih, const float* __restrict__ whh,
                            const float* __restrict__ bih, const float* __restrict__ bhh,
                            float* __restrict__ g, int mode) {
  int t = blockIdx.x * blockDim.x + threadIdx.x;   // B*256
  int b = t >> 8, j = t & 255;
  float acc = bih[j] + bhh[j];
  if (mode == 1) {
    const float* w  = wih + (size_t)j * 128;
    const float* qa = xa + (size_t)b * 64;
    const float* qb = xb + (size_t)b * 64;
#pragma unroll 4
    for (int k = 0; k < 64; ++k) acc += qa[k] * w[k];
#pragma unroll 4
    for (int k = 0; k < 64; ++k) acc += qb[k] * w[64 + k];
  } else if (mode == 0) {
    const float* w  = wih + (size_t)j * 64;
    const float* qa = xa + (size_t)b * 64;
#pragma unroll 4
    for (int k = 0; k < 64; ++k) acc += qa[k] * w[k];
  }
  const float* wh = whh + (size_t)j * 64;
  const float* hb = h + (size_t)b * 64;
#pragma unroll 4
  for (int k = 0; k < 64; ++k) acc += hb[k] * wh[k];
  g[t] = acc;
}

__global__ void k_lstm_gate(const float* __restrict__ g, float* __restrict__ h,
                            float* __restrict__ c, float* __restrict__ qout) {
  int t = blockIdx.x * blockDim.x + threadIdx.x;   // B*64
  int b = t >> 6, k = t & 63;
  const float* gb = g + (size_t)b * 256;
  float gi = gb[k], gf = gb[64 + k], gg = gb[128 + k], go = gb[192 + k];
  float c2 = sigm(gf) * c[t] + sigm(gi) * tanhf(gg);
  float h2 = sigm(go) * tanhf(c2);
  c[t] = c2;
  h[t] = h2;
  if (qout) qout[t] = h2;
}

__global__ void k_attn_init(unsigned* __restrict__ emax_u, float* __restrict__ esum,
                            float* __restrict__ readout) {
  int t = blockIdx.x * blockDim.x + threadIdx.x;   // B*64
  readout[t] = 0.0f;
  if (t < BAT) { emax_u[t] = 0u; esum[t] = 0.0f; }
}

__global__ void k_attn_e(const float* __restrict__ out, const float* __restrict__ q,
                         const int* __restrict__ ng, float* __restrict__ e_arr,
                         unsigned* __restrict__ emax_u) {
  int n = blockIdx.x * blockDim.x + threadIdx.x;
  if (n >= NN) return;
  int s = ng[n];
  const float* orow = out + (size_t)n * 64;
  const float* qrow = q + (size_t)s * 64;
  float acc = 0.0f;
#pragma unroll 4
  for (int k = 0; k < 64; ++k) acc += orow[k] * qrow[k];
  e_arr[n] = acc;
  atomicMax(emax_u + s, fenc(acc));
}

__global__ void k_attn_exp(const float* __restrict__ e_arr, const int* __restrict__ ng,
                           const unsigned* __restrict__ emax_u, float* __restrict__ ee,
                           float* __restrict__ esum) {
  int n = blockIdx.x * blockDim.x + threadIdx.x;
  if (n >= NN) return;
  int s = ng[n];
  float v = expf(e_arr[n] - fdec(emax_u[s]));
  ee[n] = v;
  atomicAdd(esum + s, v);
}

__global__ void k_attn_read(const float* __restrict__ out, const float* __restrict__ ee,
                            const float* __restrict__ esum, const int* __restrict__ ng,
                            float* __restrict__ readout) {
  int t = blockIdx.x * blockDim.x + threadIdx.x;   // N*64
  int n = t >> 6, f = t & 63;
  int s = ng[n];
  float alpha = ee[n] / esum[s];
  atomicAdd(readout + (size_t)s * 64 + f, out[t] * alpha);
}

__global__ void k_final(const float* __restrict__ q, const float* __restrict__ readout,
                        const float* __restrict__ w3, const float* __restrict__ b3,
                        const float* __restrict__ pw, const float* __restrict__ pb,
                        float* __restrict__ pred) {
  int b = threadIdx.x;   // 256 threads, 1 block
  float x[128];
#pragma unroll 4
  for (int k = 0; k < 64; ++k) {
    x[k] = q[(size_t)b * 64 + k];
    x[64 + k] = readout[(size_t)b * 64 + k];
  }
  float acc = pb[0];
  for (int j = 0; j < 128; ++j) {
    float a = b3[j];
    const float* wr = w3 + (size_t)j * 128;
#pragma unroll 4
    for (int k = 0; k < 128; ++k) a += x[k] * wr[k];
    acc += fmaxf(a, 0.0f) * pw[j];
  }
  pred[b] = acc;
}

// ---------------------------------------------------------------------------
// Orchestration
// ---------------------------------------------------------------------------
extern "C" void kernel_launch(void* const* d_in, const int* in_sizes, int n_in,
                              void* d_out, int out_size, void* d_ws, size_t ws_size,
                              hipStream_t stream) {
  const float* n_feat  = (const float*)d_in[0];
  const float* e_feat  = (const float*)d_in[1];
  const int*   src     = (const int*)d_in[2];
  const int*   dst     = (const int*)d_in[3];
  const int*   ngraph  = (const int*)d_in[4];
  const float* lin0_w  = (const float*)d_in[6];
  const float* lin0_b  = (const float*)d_in[7];
  const float* en_w1   = (const float*)d_in[8];
  const float* en_b1   = (const float*)d_in[9];
  const float* en_w2   = (const float*)d_in[10];
  const float* en_b2   = (const float*)d_in[11];
  const float* conv_b  = (const float*)d_in[12];
  const float* gru_wih = (const float*)d_in[13];
  const float* gru_whh = (const float*)d_in[14];
  const float* gru_bih = (const float*)d_in[15];
  const float* gru_bhh = (const float*)d_in[16];
  const float* ls_wih0  = (const float*)d_in[17];
  const float* ls_wih12 = (const float*)d_in[18];
  const float* ls_whh   = (const float*)d_in[19];
  const float* ls_bih   = (const float*)d_in[20];
  const float* ls_bhh   = (const float*)d_in[21];
  const float* lin3_w  = (const float*)d_in[22];
  const float* lin3_b  = (const float*)d_in[23];
  const float* pred_w  = (const float*)d_in[24];
  const float* pred_b  = (const float*)d_in[25];

  float* out_h = (float*)d_out;                    // N*64 fp32: h == out (final output)
  float* pred  = (float*)d_out + (size_t)NN * D;   // 256 predictions

  // workspace carve-up (~862 MB total; bf16 ew dominates)
  char* ws = (char*)d_ws;
  auto alloc = [&](size_t bytes) -> char* {
    char* p = ws;
    ws += (bytes + 255) & ~(size_t)255;
    return p;
  };
  __bf16* ewt    = (__bf16*)alloc((size_t)NE * 4096 * 2);   // 819.2 MB
  __bf16* he_bf  = (__bf16*)alloc((size_t)NE * 128 * 2);    // 25.6 MB
  __bf16* out_bf = (__bf16*)alloc((size_t)NN * D * 2);      // 5.1 MB
  __bf16* w2_bf  = (__bf16*)alloc((size_t)4096 * 128 * 2);
  __bf16* wih_bf = (__bf16*)alloc((size_t)192 * 64 * 2);
  __bf16* whh_bf = (__bf16*)alloc((size_t)192 * 64 * 2);
  float* mbuf    = (float*)alloc((size_t)NN * D * 4);       // 10.2 MB
  float* g_lstm  = (float*)alloc((size_t)BAT * 256 * 4);
  float* hs      = (float*)alloc((size_t)3 * BAT * D * 4);
  float* cs      = (float*)alloc((size_t)3 * BAT * D * 4);
  float* qbuf    = (float*)alloc((size_t)BAT * D * 4);
  float* readout = (float*)alloc((size_t)BAT * D * 4);
  unsigned* emax = (unsigned*)alloc((size_t)BAT * 4);
  float* esum    = (float*)alloc((size_t)BAT * 4);
  float* e_arr   = (float*)alloc((size_t)NN * 4);
  float* ee      = (float*)alloc((size_t)NN * 4);

  // --- weight conversions (once) ---
  k_f32_to_bf16<<<(4096 * 128 + 255) / 256, 256, 0, stream>>>(en_w2, w2_bf, 4096 * 128);
  k_f32_to_bf16<<<(192 * 64 + 255) / 256, 256, 0, stream>>>(gru_wih, wih_bf, 192 * 64);
  k_f32_to_bf16<<<(192 * 64 + 255) / 256, 256, 0, stream>>>(gru_whh, whh_bf, 192 * 64);

  // --- input projection (out = h) ---
  k_lin0<<<(NN * D) / 256, 256, 0, stream>>>(n_feat, lin0_w, lin0_b, out_h, out_bf);

  // --- edge network: bf16 WMMA GEMM, stored transposed per edge ---
  k_edge_l1<<<(NE * 128) / 256, 256, 0, stream>>>(e_feat, en_w1, en_b1, he_bf);
  k_edge_gemm<<<dim3(NE / 16, 256 / 4), 128, 0, stream>>>(he_bf, w2_bf, en_b2, ewt);

  // --- 6 message-passing + GRU steps ---
  for (int step = 0; step < 6; ++step) {
    k_msg_init<<<(NN * D) / 256, 256, 0, stream>>>(conv_b, mbuf);
    k_message<<<5000, 64, 0, stream>>>(out_bf, ewt, src, dst, mbuf);
    k_gru<<<(NN / 16 + 7) / 8, 256, 0, stream>>>(mbuf, out_h, out_bf, wih_bf, whh_bf,
                                                 gru_bih, gru_bhh, NN / 16);
  }

  // --- Set2Set pooling ---
  k_zero_f32<<<(6 * BAT * D + 255) / 256, 256, 0, stream>>>(hs, 6 * BAT * D); // hs+cs contiguous
  for (int it = 0; it < 6; ++it) {
    // layer 0: x = q_star (zeros at it==0)
    k_lstm_gemm<<<BAT, 256, 0, stream>>>(qbuf, readout, hs + 0 * BAT * D,
                                         ls_wih0, ls_whh + 0 * 256 * 64,
                                         ls_bih + 0 * 256, ls_bhh + 0 * 256,
                                         g_lstm, it == 0 ? 2 : 1);
    k_lstm_gate<<<BAT * D / 256, 256, 0, stream>>>(g_lstm, hs + 0 * BAT * D, cs + 0 * BAT * D, (float*)nullptr);
    // layer 1: x = hs[0]
    k_lstm_gemm<<<BAT, 256, 0, stream>>>(hs + 0 * BAT * D, (const float*)nullptr, hs + 1 * BAT * D,
                                         ls_wih12 + 0 * 256 * 64, ls_whh + 1 * 256 * 64,
                                         ls_bih + 1 * 256, ls_bhh + 1 * 256, g_lstm, 0);
    k_lstm_gate<<<BAT * D / 256, 256, 0, stream>>>(g_lstm, hs + 1 * BAT * D, cs + 1 * BAT * D, (float*)nullptr);
    // layer 2: x = hs[1]; q = hs[2]
    k_lstm_gemm<<<BAT, 256, 0, stream>>>(hs + 1 * BAT * D, (const float*)nullptr, hs + 2 * BAT * D,
                                         ls_wih12 + 1 * 256 * 64, ls_whh + 2 * 256 * 64,
                                         ls_bih + 2 * 256, ls_bhh + 2 * 256, g_lstm, 0);
    k_lstm_gate<<<BAT * D / 256, 256, 0, stream>>>(g_lstm, hs + 2 * BAT * D, cs + 2 * BAT * D, qbuf);
    // attention softmax over nodes + weighted readout
    k_attn_init<<<BAT * D / 256, 256, 0, stream>>>(emax, esum, readout);
    k_attn_e<<<(NN + 255) / 256, 256, 0, stream>>>(out_h, qbuf, ngraph, e_arr, emax);
    k_attn_exp<<<(NN + 255) / 256, 256, 0, stream>>>(e_arr, ngraph, emax, ee, esum);
    k_attn_read<<<(NN * D) / 256, 256, 0, stream>>>(out_h, ee, esum, ngraph, readout);
  }

  // --- final MLP: y = relu(q_star @ lin3^T + b); pred = y @ pred_w^T + b ---
  k_final<<<1, BAT, 0, stream>>>(qbuf, readout, lin3_w, lin3_b, pred_w, pred_b, pred);
}
